// MotionGenerator_8589935364
// MI455X (gfx1250) — matrix operand order repaired and checked by reading, not compiled, not run on hardware
//
#include <hip/hip_runtime.h>
#include <hip/hip_bf16.h>
#include <math.h>

typedef __attribute__((ext_vector_type(16))) _Float16 v16h;
typedef __attribute__((ext_vector_type(8)))  float    v8f;

#define Bsz  2048
#define Tt   60
#define DLAT 100
#define Dd   512
#define Hh   256
#define Jj   6

#define WMMA_F16(acc, a, b) \
  __builtin_amdgcn_wmma_f32_16x16x32_f16(false, (a), false, (b), (short)0, (acc), false, false)

#if defined(__gfx1250__) && __has_builtin(__builtin_amdgcn_tensor_load_to_lds) && \
    __has_builtin(__builtin_amdgcn_s_wait_tensorcnt)
#define USE_TDM 1
#else
#define USE_TDM 0
#endif

// ---------- fixed kinematic constants ----------
__constant__ float c_axes[6][3] = {{0,0,1},{1,0,0},{0,1,0},{1,0,0},{0,1,0},{1,0,0}};
__constant__ float c_offs[6][3] = {{0,0,0.10f},{0,0,0},{0,0,0},{0,-0.25f,0},{0,0,0},{0,-0.25f,0}};
__constant__ float c_lower[6] = {-3.1416f,-1.5708f,-3.1416f,-2.6f,-1.5708f,-1.2f};
__constant__ float c_upper[6] = { 3.1416f, 1.5708f, 3.1416f, 0.1f, 1.5708f, 1.2f};
__constant__ float c_f1[3] = {0.0f,-0.08f, 0.02f};
__constant__ float c_f4[3] = {0.0f,-0.08f,-0.02f};

// ---------- TDM: DMA one contiguous 16KB (16x512 half) tile global -> LDS ----------
#if USE_TDM
__device__ inline void tdm_load_16x512(const _Float16* gsrc, void* lds_dst) {
  typedef __attribute__((ext_vector_type(4))) unsigned int u32x4;
  typedef __attribute__((ext_vector_type(8))) int i32x8;
  typedef __attribute__((ext_vector_type(4))) int i32x4;
  unsigned lds_off = (unsigned)(uintptr_t)lds_dst;       // addr[31:0] == LDS offset
  unsigned long long ga = (unsigned long long)(uintptr_t)gsrc;
  const unsigned nel = 16u * 512u;                       // 8192 elements, data_size=2B
  // D# group0: count=1 | lds_addr | global_addr[56:0] | type=2
  u32x4 g0 = { 1u,
               lds_off,
               (unsigned)(ga & 0xFFFFFFFFu),
               (unsigned)((ga >> 32) & 0x01FFFFFFu) | (2u << 30) };
  // D# group1: data_size=2B; tensor_dim0=nel; tensor_dim1=1; tile_dim0=nel; dim0_stride=nel
  i32x8 g1 = { (int)0x00010000,                          // workgroup_mask=0, data_size=1(2B)
               (int)((nel & 0xFFFFu) << 16),             // tensor_dim0[15:0] (bits 63:48)
               (int)(((nel >> 16) & 0xFFFFu) | (1u << 16)), // tensor_dim0[31:16] | tensor_dim1=1
               (int)(nel << 16),                         // tensor_dim1[31:16]=0 | tile_dim0
               0,                                        // tile_dim1=0, tile_dim2=0 (unused)
               (int)nel,                                 // tensor_dim0_stride[31:0]
               0, 0 };
  i32x4 z4 = {0, 0, 0, 0};
  i32x8 z8 = {0, 0, 0, 0, 0, 0, 0, 0};
  __builtin_amdgcn_tensor_load_to_lds(g0, g1, z4, z4, z8, 0);   // 6-arg form (clang-23)
  __builtin_amdgcn_s_wait_tensorcnt(0);
}
#endif

// ---------- WMMA fragment helpers (16x16x32 f16, wave32) ----------
// A fragment: 16x32 (MxK) row-major source with leading dim ld (halfs).
// Per ISA 7.12.2: lanes 0-15 hold M=0..15, K in {0..7,16..23}; lanes 16-31 hold K in {8..15,24..31}.
__device__ inline v16h load_a_frag(const _Float16* A, int ld, int kc) {
  int lane = threadIdx.x & 31;
  int m = lane & 15;
  int kb = (lane >> 4) << 3;                 // 0 or 8
  const _Float16* p = A + (size_t)m * ld + kc * 32;
  v16h a;
#pragma unroll
  for (int e = 0; e < 8; ++e) { a[e] = p[kb + e]; a[8 + e] = p[16 + kb + e]; }
  return a;
}

// B fragment: 32x16 (KxN). B[k][n] = W[n0+n][k] (W row-major [N][K], ld = ldk).
// Lanes 0-15 hold K in {0..7,16..23} of column n=lane; lanes 16-31 hold K in {8..15,24..31}.
__device__ inline v16h load_b_frag(const _Float16* W, int ldk, int n0, int kc) {
  int lane = threadIdx.x & 31;
  int n = lane & 15;
  int koff = (lane >> 4) << 3;               // 0 or 8
  const _Float16* p = W + (size_t)(n0 + n) * ldk + kc * 32;
  v16h b;
#pragma unroll
  for (int e = 0; e < 8; ++e) { b[e] = p[koff + e]; b[8 + e] = p[16 + koff + e]; }
  return b;
}

// ---------- 1: fc GEMM (K=100, scalar; tiny) ----------
__global__ __launch_bounds__(256)
void fc_kernel(const float* __restrict__ z, const float* __restrict__ Wfc,
               const float* __restrict__ bfc, float* __restrict__ h0) {
  int idx = blockIdx.x * 256 + threadIdx.x;
  if (idx >= Bsz * Dd) return;
  int b = idx >> 9, d = idx & 511;
  const float* zr = z + (size_t)b * DLAT;
  const float* wr = Wfc + (size_t)d * DLAT;
  float s = bfc[d];
#pragma unroll 4
  for (int k = 0; k < DLAT; ++k) s = fmaf(zr[k], wr[k], s);
  h0[idx] = s;
}

// ---------- 2: batchnorm stats (per channel over B) ----------
__global__ __launch_bounds__(256)
void bn_stats_kernel(const float* __restrict__ h0, float* __restrict__ stats) {
  __shared__ float s1[256], s2[256];
  int d = blockIdx.x, tid = threadIdx.x;
  float a = 0.f, b = 0.f;
  for (int i = tid; i < Bsz; i += 256) {
    float x = h0[(size_t)i * Dd + d];
    a += x; b += x * x;
  }
  s1[tid] = a; s2[tid] = b; __syncthreads();
  for (int s = 128; s > 0; s >>= 1) {
    if (tid < s) { s1[tid] += s1[tid + s]; s2[tid] += s2[tid + s]; }
    __syncthreads();
  }
  if (tid == 0) {
    float mu  = s1[0] * (1.f / Bsz);
    float var = s2[0] * (1.f / Bsz) - mu * mu;
    stats[d] = mu;
    stats[Dd + d] = rsqrtf(var + 1e-5f);
  }
}

// ---------- 3: BN apply + LeakyReLU -> f16 ----------
__global__ __launch_bounds__(256)
void bn_apply_kernel(const float* __restrict__ h0, const float* __restrict__ stats,
                     const float* __restrict__ g, const float* __restrict__ bb,
                     _Float16* __restrict__ hh) {
  int idx = blockIdx.x * 256 + threadIdx.x;
  if (idx >= Bsz * Dd) return;
  int d = idx & 511;
  float x = (h0[idx] - stats[d]) * stats[Dd + d] * g[d] + bb[d];
  x = (x >= 0.f) ? x : 0.2f * x;
  hh[idx] = (_Float16)x;
}

// ---------- 4: f32 -> f16 convert ----------
__global__ __launch_bounds__(256)
void f2h_kernel(const float* __restrict__ s, _Float16* __restrict__ d, int n) {
  int i = blockIdx.x * 256 + threadIdx.x;
  if (i < n) d[i] = (_Float16)s[i];
}

// ---------- 5: xg = h @ [Wih_f;Wih_b]^T + bias  (WMMA, M=2048,N=1536,K=512) ----------
__global__ __launch_bounds__(256)
void xg_kernel(const _Float16* __restrict__ hh, const _Float16* __restrict__ Wih_h,
               const float* __restrict__ bih_f, const float* __restrict__ bih_b,
               float* __restrict__ xg) {
  __shared__ _Float16 a_s[16][Dd];
  int b0 = blockIdx.x * 16;
  int tid = threadIdx.x, wave = tid >> 5;
#if USE_TDM
  if (tid < 32) tdm_load_16x512(hh + (size_t)b0 * Dd, &a_s[0][0]);   // one TDM op from wave 0
#else
  for (int i = tid; i < 16 * Dd; i += 256)
    (&a_s[0][0])[i] = hh[(size_t)(b0 + (i >> 9)) * Dd + (i & 511)];
#endif
  __syncthreads();
  int n0 = blockIdx.y * 128 + wave * 16;
  v8f acc = {};
#pragma unroll
  for (int kc = 0; kc < 16; ++kc) {
    v16h a = load_a_frag(&a_s[0][0], Dd, kc);
    v16h b = load_b_frag(Wih_h, Dd, n0, kc);
    acc = WMMA_F16(acc, a, b);
  }
  int lane = tid & 31, n = lane & 15, mb = (lane >> 4) * 8;
  int col = n0 + n;
  float bias = (col < 768) ? bih_f[col] : bih_b[col - 768];
#pragma unroll
  for (int v = 0; v < 8; ++v)
    xg[(size_t)(b0 + mb + v) * 1536 + col] = acc[v] + bias;
}

// ---------- 6: bidirectional GRU scan (WMMA per step, LDS-resident hidden) ----------
__global__ __launch_bounds__(256)
void gru_kernel(const float* __restrict__ xg, const _Float16* __restrict__ Whh_h,
                const float* __restrict__ bhh_f, const float* __restrict__ bhh_b,
                _Float16* __restrict__ gru_out) {
  __shared__ _Float16 h_s[16][Hh];        // 8 KB
  __shared__ float gh_s[16][3 * Hh];      // 48 KB
  int b0 = blockIdx.x * 16;
  int dir = blockIdx.y;                    // 0=fwd, 1=bwd
  int tid = threadIdx.x, wave = tid >> 5;
  const float* bhh = dir ? bhh_b : bhh_f;
  const _Float16* Whh = Whh_h + (size_t)dir * 768 * Hh;     // [768][256] half
  const float* xg_base = xg + (size_t)b0 * 1536 + dir * 768;

  // prefetch this wave's Whh column block (global_prefetch_b8)
  __builtin_prefetch(Whh + (size_t)(wave * 96) * Hh, 0, 1);

  for (int i = tid; i < 16 * Hh; i += 256) (&h_s[0][0])[i] = (_Float16)0.f;

  int lane = tid & 31, nlo = lane & 15, mb = (lane >> 4) * 8;
  int gm = tid >> 4;                 // gate row 0..15
  int gc0 = (tid & 15) * 16;         // gate col base 0..240

  // x is broadcast over T -> the gate inputs are loop-invariant: register-cache them.
  const float* xr = xg_base + (size_t)gm * 1536;
  float xr_r[16], xr_z[16], xr_n[16];
#pragma unroll
  for (int i = 0; i < 16; ++i) {
    int c = gc0 + i;
    xr_r[i] = xr[c];
    xr_z[i] = xr[Hh + c];
    xr_n[i] = xr[2 * Hh + c];
  }
  __syncthreads();

  for (int step = 0; step < Tt; ++step) {
    // gh = h @ Whh^T + bhh : [16,256] x [256,768]
#pragma unroll
    for (int nt = 0; nt < 6; ++nt) {
      int n0 = wave * 96 + nt * 16;
      v8f acc = {};
#pragma unroll
      for (int kc = 0; kc < 8; ++kc) {
        v16h a = load_a_frag(&h_s[0][0], Hh, kc);
        v16h b = load_b_frag(Whh, Hh, n0, kc);
        acc = WMMA_F16(acc, a, b);
      }
      float bias = bhh[n0 + nlo];
#pragma unroll
      for (int v = 0; v < 8; ++v) gh_s[mb + v][n0 + nlo] = acc[v] + bias;
    }
    __syncthreads();

    // gates + state update
    int tstore = dir ? (Tt - 1 - step) : step;
    _Float16* outrow = gru_out + ((size_t)(b0 + gm) * Tt + tstore) * (2 * Hh) + dir * Hh;
#pragma unroll
    for (int i = 0; i < 16; ++i) {
      int c = gc0 + i;
      float gr = xr_r[i] + gh_s[gm][c];
      float gz = xr_z[i] + gh_s[gm][Hh + c];
      float hn = gh_s[gm][2 * Hh + c];
      float r  = 1.f / (1.f + __expf(-gr));
      float zz = 1.f / (1.f + __expf(-gz));
      float nn = tanhf(xr_n[i] + r * hn);
      float hp = (float)h_s[gm][c];
      float hv = (1.f - zz) * nn + zz * hp;
      h_s[gm][c] = (_Float16)hv;
      outrow[c]  = (_Float16)hv;
    }
    __syncthreads();
  }
}

// ---------- 7: attention scores s[b,t] = tanh(gru @ W1^T + b1) . W2  (WMMA) ----------
__global__ __launch_bounds__(256)
void attn_score_kernel(const _Float16* __restrict__ gru_out, const _Float16* __restrict__ W1_h,
                       const float* __restrict__ b1, const float* __restrict__ W2,
                       float* __restrict__ scores) {
  __shared__ _Float16 a_s[16][2 * Hh];
  __shared__ float s_acc[16];
  int r0 = blockIdx.x * 16;       // row in B*T
  int tid = threadIdx.x, wave = tid >> 5;
#if USE_TDM
  if (tid < 32) tdm_load_16x512(gru_out + (size_t)r0 * (2 * Hh), &a_s[0][0]);
#else
  for (int i = tid; i < 16 * 2 * Hh; i += 256)
    (&a_s[0][0])[i] = gru_out[(size_t)(r0 + (i >> 9)) * (2 * Hh) + (i & 511)];
#endif
  if (tid < 16) s_acc[tid] = 0.f;
  __syncthreads();
  int lane = tid & 31, n = lane & 15, mb = (lane >> 4) * 8;
#pragma unroll
  for (int nt = 0; nt < 2; ++nt) {
    int n0 = wave * 32 + nt * 16;
    v8f acc = {};
#pragma unroll
    for (int kc = 0; kc < 16; ++kc) {
      v16h a = load_a_frag(&a_s[0][0], 2 * Hh, kc);
      v16h b = load_b_frag(W1_h, 2 * Hh, n0, kc);
      acc = WMMA_F16(acc, a, b);
    }
    float bb = b1[n0 + n], ww = W2[n0 + n];
#pragma unroll
    for (int v = 0; v < 8; ++v)
      atomicAdd(&s_acc[mb + v], tanhf(acc[v] + bb) * ww);   // ds_add_f32
  }
  __syncthreads();
  if (tid < 16) scores[r0 + tid] = s_acc[tid];
}

// ---------- 8: softmax over T + context c[b,512] ----------
__global__ __launch_bounds__(256)
void ctx_kernel(const float* __restrict__ scores, const _Float16* __restrict__ gru_out,
                float* __restrict__ ctx) {
  __shared__ float w_s[Tt];
  __shared__ float red[256];
  int b = blockIdx.x, tid = threadIdx.x;
  float v = (tid < Tt) ? scores[b * Tt + tid] : -1e30f;
  red[tid] = v; __syncthreads();
  for (int s = 128; s > 0; s >>= 1) { if (tid < s) red[tid] = fmaxf(red[tid], red[tid + s]); __syncthreads(); }
  float mx = red[0]; __syncthreads();
  float e = (tid < Tt) ? __expf(v - mx) : 0.f;
  red[tid] = e; __syncthreads();
  for (int s = 128; s > 0; s >>= 1) { if (tid < s) red[tid] += red[tid + s]; __syncthreads(); }
  float inv = 1.f / red[0];
  if (tid < Tt) w_s[tid] = e * inv;
  __syncthreads();
  for (int d = tid; d < 2 * Hh; d += 256) {
    const _Float16* g = gru_out + (size_t)b * Tt * (2 * Hh) + d;
    float c = 0.f;
#pragma unroll 4
    for (int t = 0; t < Tt; ++t) c += w_s[t] * (float)g[(size_t)t * (2 * Hh)];
    ctx[(size_t)b * (2 * Hh) + d] = c;
  }
}

// ---------- 9: joint head + forward kinematics ----------
__global__ __launch_bounds__(256)
void fk_kernel(const _Float16* __restrict__ gru_out, const float* __restrict__ ctx,
               const float* __restrict__ joint_W, const float* __restrict__ joint_b,
               float* __restrict__ out) {
  __shared__ float w_s[Jj][2 * Hh];
  int tid = threadIdx.x;
  for (int i = tid; i < Jj * 2 * Hh; i += 256) (&w_s[0][0])[i] = joint_W[i];
  __syncthreads();
  int idx = blockIdx.x * 256 + tid;
  if (idx >= Bsz * Tt) return;
  int b = idx / Tt;
  const _Float16* g = gru_out + (size_t)idx * (2 * Hh);
  const float* c = ctx + (size_t)b * (2 * Hh);

  float acc[Jj];
#pragma unroll
  for (int j = 0; j < Jj; ++j) acc[j] = joint_b[j];
  for (int d = 0; d < 2 * Hh; ++d) {
    float x = (float)g[d] + c[d];
#pragma unroll
    for (int j = 0; j < Jj; ++j) acc[j] = fmaf(x, w_s[j][d], acc[j]);
  }

  // forward kinematics (Rodrigues chain)
  float R[3][3] = {{1,0,0},{0,1,0},{0,0,1}};
  float p[3] = {0,0,0};
  float sh[3], fa[3], wr[3];
#pragma unroll
  for (int j = 0; j < Jj; ++j) {
    float th = acc[j] * (c_upper[j] - c_lower[j]) + c_lower[j];
    float s = __sinf(th), co = __cosf(th);
    const float* a = c_axes[j];
    float K[3][3] = {{0,-a[2],a[1]},{a[2],0,-a[0]},{-a[1],a[0],0}};
    float K2[3][3], Rj[3][3], Rn[3][3];
#pragma unroll
    for (int r = 0; r < 3; ++r)
#pragma unroll
      for (int cc = 0; cc < 3; ++cc)
        K2[r][cc] = K[r][0]*K[0][cc] + K[r][1]*K[1][cc] + K[r][2]*K[2][cc];
#pragma unroll
    for (int r = 0; r < 3; ++r)
#pragma unroll
      for (int cc = 0; cc < 3; ++cc)
        Rj[r][cc] = ((r==cc)?1.f:0.f) + s*K[r][cc] + (1.f-co)*K2[r][cc];
#pragma unroll
    for (int r = 0; r < 3; ++r)
      p[r] += R[r][0]*c_offs[j][0] + R[r][1]*c_offs[j][1] + R[r][2]*c_offs[j][2];
#pragma unroll
    for (int r = 0; r < 3; ++r)
#pragma unroll
      for (int cc = 0; cc < 3; ++cc)
        Rn[r][cc] = R[r][0]*Rj[0][cc] + R[r][1]*Rj[1][cc] + R[r][2]*Rj[2][cc];
#pragma unroll
    for (int r = 0; r < 3; ++r)
#pragma unroll
      for (int cc = 0; cc < 3; ++cc) R[r][cc] = Rn[r][cc];
    if (j == 1) { sh[0]=p[0]; sh[1]=p[1]; sh[2]=p[2]; }
    if (j == 3) { fa[0]=p[0]; fa[1]=p[1]; fa[2]=p[2]; }
    if (j == 5) { wr[0]=p[0]; wr[1]=p[1]; wr[2]=p[2]; }
  }
  float f1[3], f4[3];
#pragma unroll
  for (int r = 0; r < 3; ++r) {
    f1[r] = wr[r] + R[r][0]*c_f1[0] + R[r][1]*c_f1[1] + R[r][2]*c_f1[2];
    f4[r] = wr[r] + R[r][0]*c_f4[0] + R[r][1]*c_f4[1] + R[r][2]*c_f4[2];
  }
  float d1 = sqrtf((sh[0]-fa[0])*(sh[0]-fa[0]) + (sh[1]-fa[1])*(sh[1]-fa[1]) + (sh[2]-fa[2])*(sh[2]-fa[2]));
  float d2 = sqrtf((wr[0]-fa[0])*(wr[0]-fa[0]) + (wr[1]-fa[1])*(wr[1]-fa[1]) + (wr[2]-fa[2])*(wr[2]-fa[2]));
  float inv = 1.f / (0.5f * (d1 + d2));
  float* o = out + (size_t)idx * 9;
#pragma unroll
  for (int r = 0; r < 3; ++r) {
    o[0*3 + r] = (wr[r] - sh[r]) * inv;
    o[1*3 + r] = (f1[r] - sh[r]) * inv;
    o[2*3 + r] = (f4[r] - sh[r]) * inv;
  }
}

// ---------- host launcher ----------
extern "C" void kernel_launch(void* const* d_in, const int* in_sizes, int n_in,
                              void* d_out, int out_size, void* d_ws, size_t ws_size,
                              hipStream_t stream) {
  (void)in_sizes; (void)n_in; (void)out_size; (void)ws_size;
  const float* z      = (const float*)d_in[0];
  const float* W_fc   = (const float*)d_in[1];
  const float* b_fc   = (const float*)d_in[2];
  const float* bn_g   = (const float*)d_in[3];
  const float* bn_b   = (const float*)d_in[4];
  const float* Wih_f  = (const float*)d_in[5];
  const float* Whh_f  = (const float*)d_in[6];
  const float* bih_f  = (const float*)d_in[7];
  const float* bhh_f  = (const float*)d_in[8];
  const float* Wih_b  = (const float*)d_in[9];
  const float* Whh_b  = (const float*)d_in[10];
  const float* bih_b  = (const float*)d_in[11];
  const float* bhh_b  = (const float*)d_in[12];
  const float* attnW1 = (const float*)d_in[13];
  const float* attnb1 = (const float*)d_in[14];
  const float* attnW2 = (const float*)d_in[15];
  const float* jointW = (const float*)d_in[16];
  const float* jointb = (const float*)d_in[17];
  float* out = (float*)d_out;

  char* w = (char*)d_ws;
  size_t off = 0;
  auto take = [&](size_t bytes) -> void* {
    off = (off + 255) & ~(size_t)255;
    void* p = w + off;
    off += bytes;
    return p;
  };
  float*     h0     = (float*)    take((size_t)Bsz * Dd * 4);
  float*     stats  = (float*)    take((size_t)2 * Dd * 4);
  _Float16*  hh     = (_Float16*) take((size_t)Bsz * Dd * 2);
  _Float16*  Wih_h  = (_Float16*) take((size_t)1536 * Dd * 2);
  _Float16*  Whh_h  = (_Float16*) take((size_t)1536 * Hh * 2);
  _Float16*  W1_h   = (_Float16*) take((size_t)Hh * Dd * 2);
  float*     xg     = (float*)    take((size_t)Bsz * 1536 * 4);
  _Float16*  gruo   = (_Float16*) take((size_t)Bsz * Tt * 2 * Hh * 2);
  float*     scores = (float*)    take((size_t)Bsz * Tt * 4);
  float*     ctx    = (float*)    take((size_t)Bsz * 2 * Hh * 4);

  // 1-3: fc + batchnorm + leaky -> f16
  fc_kernel<<<(Bsz * Dd + 255) / 256, 256, 0, stream>>>(z, W_fc, b_fc, h0);
  bn_stats_kernel<<<Dd, 256, 0, stream>>>(h0, stats);
  bn_apply_kernel<<<(Bsz * Dd + 255) / 256, 256, 0, stream>>>(h0, stats, bn_g, bn_b, hh);

  // 4: weight converts
  f2h_kernel<<<(768 * Dd + 255) / 256, 256, 0, stream>>>(Wih_f, Wih_h, 768 * Dd);
  f2h_kernel<<<(768 * Dd + 255) / 256, 256, 0, stream>>>(Wih_b, Wih_h + (size_t)768 * Dd, 768 * Dd);
  f2h_kernel<<<(768 * Hh + 255) / 256, 256, 0, stream>>>(Whh_f, Whh_h, 768 * Hh);
  f2h_kernel<<<(768 * Hh + 255) / 256, 256, 0, stream>>>(Whh_b, Whh_h + (size_t)768 * Hh, 768 * Hh);
  f2h_kernel<<<(Hh * Dd + 255) / 256, 256, 0, stream>>>(attnW1, W1_h, Hh * Dd);

  // 5: time-invariant input projection (WMMA, TDM-staged A tile)
  xg_kernel<<<dim3(Bsz / 16, 1536 / 128), 256, 0, stream>>>(hh, Wih_h, bih_f, bih_b, xg);

  // 6: bidirectional GRU scan (WMMA)
  gru_kernel<<<dim3(Bsz / 16, 2), 256, 0, stream>>>(xg, Whh_h, bhh_f, bhh_b, gruo);

  // 7-8: attention
  attn_score_kernel<<<(Bsz * Tt) / 16, 256, 0, stream>>>(gruo, W1_h, attnb1, attnW2, scores);
  ctx_kernel<<<Bsz, 256, 0, stream>>>(scores, gruo, ctx);

  // 9: joints + FK
  fk_kernel<<<(Bsz * Tt + 255) / 256, 256, 0, stream>>>(gruo, ctx, jointW, jointb, out);
}
